// FC_GNN_84421877170709
// MI455X (gfx1250) — matrix operand order repaired and checked by reading, not compiled, not run on hardware
//
#include <hip/hip_runtime.h>
#include <hip/hip_bf16.h>

// Problem constants (from reference): BS=16, N=128, IN_NF=64, H=256, EH=128, OUT_NF=64, L=4
#define BSg   16
#define Ng    128
#define INF_  64
#define HH    256
#define EHH   128
#define ONF   64
#define LL    4
#define NTOT  (BSg * Ng)   // 2048 nodes
#define JSPLIT 2           // sender-loop split factor for the edge kernel

typedef __attribute__((ext_vector_type(16))) __bf16 v16bf;
typedef __attribute__((ext_vector_type(8)))  __bf16 v8bf;
typedef __attribute__((ext_vector_type(8)))  float  v8f;

__device__ __forceinline__ float silu_f(float x) {
    return x / (1.0f + __expf(-x));
}

// Load 16 bf16 (two 16-byte chunks) into a WMMA fragment register set.
__device__ __forceinline__ v16bf ld_frag(const __bf16* p0, const __bf16* p1) {
    v16bf r;
    ((v8bf*)&r)[0] = *(const v8bf*)p0;
    ((v8bf*)&r)[1] = *(const v8bf*)p1;
    return r;
}

// ---------------------------------------------------------------------------
// elementwise f32 -> bf16 with independent row strides (used for concat too)
// ---------------------------------------------------------------------------
__global__ void k_cvt_bf16(const float* __restrict__ src, int lds,
                           __bf16* __restrict__ dst, int ldd,
                           int rows, int cols) {
    int t = blockIdx.x * blockDim.x + threadIdx.x;
    if (t >= rows * cols) return;
    int r = t / cols, c = t - r * cols;
    dst[r * ldd + c] = (__bf16)src[r * lds + c];
}

// ---------------------------------------------------------------------------
// agg = part0 + part1, converted to bf16 straight into the concat buffer
// ---------------------------------------------------------------------------
__global__ void k_combine_agg(const float* __restrict__ p0,
                              const float* __restrict__ p1,
                              __bf16* __restrict__ dst, int ldd) {
    int t = blockIdx.x * blockDim.x + threadIdx.x;
    if (t >= NTOT * EHH) return;
    int r = t / EHH, c = t - r * EHH;
    dst[r * ldd + c] = (__bf16)(p0[t] + p1[t]);
}

// ---------------------------------------------------------------------------
// f32 [K,N] row-major  ->  bf16 [N,K] (weight transpose, run once per launch)
// ---------------------------------------------------------------------------
__global__ void k_transpose_bf16(const float* __restrict__ src,
                                 __bf16* __restrict__ dst,
                                 int K, int N) {
    int t = blockIdx.x * blockDim.x + threadIdx.x;
    if (t >= K * N) return;
    int k = t / N, n = t - k * N;
    dst[n * K + k] = (__bf16)src[t];
}

// ---------------------------------------------------------------------------
// Generic WMMA GEMM: out[M,N] = act(A[M,K] @ W[K,N] + bias) (+ res)
//   A: bf16 row-major (lda), WT: bf16 [N,K] (ldw), out/res: f32 (ldo)
//   grid = (ceil(N/128), M/16), block = 256 (8 waves, 16 cols each)
// ---------------------------------------------------------------------------
__global__ void k_gemm_wmma(const __bf16* __restrict__ A, int lda,
                            const __bf16* __restrict__ WT, int ldw,
                            const float* __restrict__ bias,
                            const float* __restrict__ res,
                            float* __restrict__ out, int ldo,
                            int N, int K, int act) {
    const int lane = threadIdx.x & 31;
    const int wave = threadIdx.x >> 5;
    const int half = lane >> 4;
    const int l16  = lane & 15;
    const int col0 = blockIdx.x * 128 + wave * 16;
    if (col0 >= N) return;                 // no barriers in this kernel: safe
    const int m0 = blockIdx.y * 16;
    const int n  = col0 + l16;

    v8f c;
    const float bv = bias ? bias[n] : 0.0f;
#pragma unroll
    for (int r = 0; r < 8; ++r) c[r] = bv;

    const __bf16* arow = A  + (m0 + l16) * lda;
    const __bf16* brow = WT + n * ldw;
    for (int kb = 0; kb < K; kb += 32) {
        // A 16-bit layout: half-wave holds K = kb + half*16 + 0..15 (contiguous, matches B)
        v16bf af = ld_frag(arow + kb + half * 16, arow + kb + half * 16 + 8);
        // B 16-bit layout: half-wave holds K = kb + half*16 + 0..15 (contiguous in WT row)
        v16bf bf = ld_frag(brow + kb + half * 16, brow + kb + half * 16 + 8);
        c = __builtin_amdgcn_wmma_f32_16x16x32_bf16(false, af, false, bf,
                                                    (short)0, c, false, false);
    }

#pragma unroll
    for (int r = 0; r < 8; ++r) {
        int row = m0 + r + half * 8;       // C/D layout: VGPR r -> M = r (+8 for upper half)
        float v = c[r];
        if (act) v = silu_f(v);
        if (res) v += res[row * ldo + n];
        out[row * ldo + n] = v;
    }
}

// ---------------------------------------------------------------------------
// Fused edge MLP + segment-sum over fully-connected graphs.
//   m1(i,j) = silu(a_i + b_j)           (be1 folded into a)
//   agg_i   = sum_{j != i} silu(m1 @ we2 + be2)
// Block = 16 receivers of one graph x one half of the sender range.
// 8 waves cover the 128 output cols; we2 B-fragments live in VGPRs throughout.
// Double-buffered LDS m1 tile: the silu/exp tile build for j+1 co-executes
// with the WMMA chain for j (XDL WMMA is TRANS-class; overlaps plain VALU).
// ---------------------------------------------------------------------------
__global__ void k_edge_agg(const float* __restrict__ a,     // [NTOT,EH] (incl. be1)
                           const float* __restrict__ b,     // [NTOT,EH]
                           const float* __restrict__ be2,   // [EH]
                           const __bf16* __restrict__ we2T, // [EH,EH] = [N,K]
                           float* __restrict__ aggp) {      // [JSPLIT,NTOT,EH]
    __shared__ __align__(16) __bf16 As[2][16 * EHH];        // 2 x 4 KB m1 tiles

    const int tid    = threadIdx.x;
    const int lane   = tid & 31;
    const int wave   = tid >> 5;
    const int half   = lane >> 4;
    const int l16    = lane & 15;
    const int tile   = blockIdx.x;        // 0..127 (8 tiles per graph)
    const int jsplit = blockIdx.y;        // 0..JSPLIT-1
    const int i0     = tile * 16;         // first receiver node (global index)
    const int jbase  = (tile >> 3) * Ng;  // first sender node of this graph
    const int j0     = jsplit * (Ng / JSPLIT);
    const int col0   = wave * 16;
    const int n      = col0 + l16;
    const int kcol   = tid & (EHH - 1);   // this thread's k within a tile row

    // Preload the full K=128 reduction worth of we2 B-fragments (32 VGPRs).
    v16bf bfrag[4];
    {
        const __bf16* brow = we2T + n * EHH;
#pragma unroll
        for (int s = 0; s < 4; ++s) {
            int kb = s * 32 + half * 16;
            bfrag[s] = ld_frag(brow + kb, brow + kb + 8);
        }
    }
    const float be2n = be2[n];

    // Per-thread slice of receiver features a[i0..i0+15][:]: loop invariant.
    float aval[8];
#pragma unroll
    for (int e = 0; e < 8; ++e)
        aval[e] = a[i0 * EHH + tid + e * 256];   // flat = m*EH + k

    // Build the 16xEH bf16 m1 tile for sender jg into buffer `buf`.
    auto build = [&](int jg, int buf) {
        const int j = jbase + jg;
        if (jg + 4 < Ng)  // warm WGP$ for an upcoming sender row
            __builtin_prefetch(b + (j + 4) * EHH + kcol, 0, 0);
#pragma unroll
        for (int e = 0; e < 8; ++e) {
            int flat = tid + e * 256;               // (m,k): flat = m*EH + k
            As[buf][flat] = (__bf16)silu_f(aval[e] + b[j * EHH + (flat & (EHH - 1))]);
        }
    };

    float acc[8];
#pragma unroll
    for (int r = 0; r < 8; ++r) acc[r] = 0.0f;

    build(j0, 0);
    __syncthreads();

    const int iters = Ng / JSPLIT;
    for (int t = 0; t < iters; ++t) {
        const int jg = j0 + t;
        const int j  = jbase + jg;
        const __bf16* arow = As[t & 1] + l16 * EHH;

        // Start the next tile build before consuming this one (other buffer).
        if (t + 1 < iters) build(jg + 1, (t + 1) & 1);

        v8f c;
#pragma unroll
        for (int r = 0; r < 8; ++r) c[r] = be2n;
#pragma unroll
        for (int s = 0; s < 4; ++s) {
            int kb = s * 32;
            v16bf af = ld_frag(arow + kb + half * 16, arow + kb + half * 16 + 8);
            c = __builtin_amdgcn_wmma_f32_16x16x32_bf16(false, af, false, bfrag[s],
                                                        (short)0, c, false, false);
        }

        // silu + self-edge mask + register accumulation of the segment sum.
#pragma unroll
        for (int r = 0; r < 8; ++r) {
            int i = i0 + r + half * 8;
            float v = silu_f(c[r]);
            acc[r] += (i != j) ? v : 0.0f;
        }
        __syncthreads();   // next build target = buffer we just consumed
    }

    float* out = aggp + (size_t)jsplit * NTOT * EHH;
#pragma unroll
    for (int r = 0; r < 8; ++r)
        out[(i0 + r + half * 8) * EHH + n] = acc[r];   // C = 1.0
}

// ---------------------------------------------------------------------------
extern "C" void kernel_launch(void* const* d_in, const int* in_sizes, int n_in,
                              void* d_out, int out_size, void* d_ws, size_t ws_size,
                              hipStream_t stream) {
    (void)in_sizes; (void)n_in; (void)out_size; (void)ws_size;

    const float* h     = (const float*)d_in[0];
    // d_in[1]=rows, d_in[2]=cols: the FC structure they encode is exploited directly
    const float* w_in  = (const float*)d_in[3];
    const float* b_in  = (const float*)d_in[4];
    const float* w_out = (const float*)d_in[5];
    const float* b_out = (const float*)d_in[6];
    const float* we1   = (const float*)d_in[7];
    const float* be1   = (const float*)d_in[8];
    const float* we2   = (const float*)d_in[9];
    const float* be2   = (const float*)d_in[10];
    const float* wn1   = (const float*)d_in[11];
    const float* bn1   = (const float*)d_in[12];
    const float* wn2   = (const float*)d_in[13];
    const float* bn2   = (const float*)d_in[14];

    char* p = (char*)d_ws;
    auto alloc = [&](size_t bytes) { char* r = p; p += (bytes + 255) & ~(size_t)255; return r; };
    float*  x      = (float*) alloc((size_t)NTOT * HH  * 4);
    float*  u      = (float*) alloc((size_t)NTOT * HH  * 4);
    float*  abuf   = (float*) alloc((size_t)NTOT * EHH * 4);
    float*  bbuf   = (float*) alloc((size_t)NTOT * EHH * 4);
    float*  aggp   = (float*) alloc((size_t)JSPLIT * NTOT * EHH * 4);
    __bf16* h_bf   = (__bf16*)alloc((size_t)NTOT * INF_ * 2);
    __bf16* x_bf   = (__bf16*)alloc((size_t)NTOT * HH  * 2);
    __bf16* xa_bf  = (__bf16*)alloc((size_t)NTOT * (HH + EHH) * 2);
    __bf16* u_bf   = (__bf16*)alloc((size_t)NTOT * HH  * 2);
    __bf16* w_inT  = (__bf16*)alloc((size_t)INF_ * HH * 2);
    __bf16* w_outT = (__bf16*)alloc((size_t)HH * ONF * 2);
    __bf16* we1T   = (__bf16*)alloc((size_t)LL * EHH * (2 * HH) * 2); // [EH, 512]
    __bf16* we2T   = (__bf16*)alloc((size_t)LL * EHH * EHH * 2);
    __bf16* wn1T   = (__bf16*)alloc((size_t)LL * HH * (HH + EHH) * 2);
    __bf16* wn2T   = (__bf16*)alloc((size_t)LL * HH * HH * 2);

    dim3 blk(256);
    auto g1 = [](int total) { return dim3((unsigned)((total + 255) / 256)); };

    // --- one-time conversions / weight transposes (bf16, [N,K] layout) ---
    hipLaunchKernelGGL(k_cvt_bf16, g1(NTOT * INF_), blk, 0, stream,
                       h, INF_, h_bf, INF_, NTOT, INF_);
    hipLaunchKernelGGL(k_transpose_bf16, g1(INF_ * HH), blk, 0, stream, w_in, w_inT, INF_, HH);
    hipLaunchKernelGGL(k_transpose_bf16, g1(HH * ONF), blk, 0, stream, w_out, w_outT, HH, ONF);
    for (int l = 0; l < LL; ++l) {
        hipLaunchKernelGGL(k_transpose_bf16, g1(2 * HH * EHH), blk, 0, stream,
                           we1 + (size_t)l * 2 * HH * EHH, we1T + (size_t)l * EHH * 2 * HH, 2 * HH, EHH);
        hipLaunchKernelGGL(k_transpose_bf16, g1(EHH * EHH), blk, 0, stream,
                           we2 + (size_t)l * EHH * EHH, we2T + (size_t)l * EHH * EHH, EHH, EHH);
        hipLaunchKernelGGL(k_transpose_bf16, g1((HH + EHH) * HH), blk, 0, stream,
                           wn1 + (size_t)l * (HH + EHH) * HH, wn1T + (size_t)l * HH * (HH + EHH), HH + EHH, HH);
        hipLaunchKernelGGL(k_transpose_bf16, g1(HH * HH), blk, 0, stream,
                           wn2 + (size_t)l * HH * HH, wn2T + (size_t)l * HH * HH, HH, HH);
    }

    // --- x = h @ w_in + b_in ---
    hipLaunchKernelGGL(k_gemm_wmma, dim3(2, NTOT / 16), blk, 0, stream,
                       h_bf, INF_, w_inT, INF_, b_in, (const float*)nullptr,
                       x, HH, HH, INF_, 0);

    for (int l = 0; l < LL; ++l) {
        hipLaunchKernelGGL(k_cvt_bf16, g1(NTOT * HH), blk, 0, stream,
                           x, HH, x_bf, HH, NTOT, HH);
        // a = x @ we1_top + be1 ; b = x @ we1_bot   (edge-GEMM decomposition)
        hipLaunchKernelGGL(k_gemm_wmma, dim3(1, NTOT / 16), blk, 0, stream,
                           x_bf, HH, we1T + (size_t)l * EHH * 2 * HH, 2 * HH,
                           be1 + (size_t)l * EHH, (const float*)nullptr,
                           abuf, EHH, EHH, HH, 0);
        hipLaunchKernelGGL(k_gemm_wmma, dim3(1, NTOT / 16), blk, 0, stream,
                           x_bf, HH, we1T + (size_t)l * EHH * 2 * HH + HH, 2 * HH,
                           (const float*)nullptr, (const float*)nullptr,
                           bbuf, EHH, EHH, HH, 0);
        // fused: silu -> (m1 @ we2 + be2) -> silu -> masked segment-sum
        hipLaunchKernelGGL(k_edge_agg, dim3(NTOT / 16, JSPLIT), blk, 0, stream,
                           abuf, bbuf, be2 + (size_t)l * EHH,
                           we2T + (size_t)l * EHH * EHH, aggp);
        // xa = concat(x, agg) in bf16 (partial-sum combine fused into concat)
        hipLaunchKernelGGL(k_cvt_bf16, g1(NTOT * HH), blk, 0, stream,
                           x, HH, xa_bf, HH + EHH, NTOT, HH);
        hipLaunchKernelGGL(k_combine_agg, g1(NTOT * EHH), blk, 0, stream,
                           aggp, aggp + (size_t)NTOT * EHH, xa_bf + HH, HH + EHH);
        // u = silu(xa @ wn1 + bn1)
        hipLaunchKernelGGL(k_gemm_wmma, dim3(2, NTOT / 16), blk, 0, stream,
                           xa_bf, HH + EHH, wn1T + (size_t)l * HH * (HH + EHH), HH + EHH,
                           bn1 + (size_t)l * HH, (const float*)nullptr,
                           u, HH, HH, HH + EHH, 1);
        hipLaunchKernelGGL(k_cvt_bf16, g1(NTOT * HH), blk, 0, stream,
                           u, HH, u_bf, HH, NTOT, HH);
        // x = x + (u @ wn2 + bn2)   (residual fused; per-element in-place safe)
        hipLaunchKernelGGL(k_gemm_wmma, dim3(2, NTOT / 16), blk, 0, stream,
                           u_bf, HH, wn2T + (size_t)l * HH * HH, HH,
                           bn2 + (size_t)l * HH, x, x, HH, HH, HH, 0);
    }

    // --- out = x @ w_out + b_out ---
    hipLaunchKernelGGL(k_cvt_bf16, g1(NTOT * HH), blk, 0, stream,
                       x, HH, x_bf, HH, NTOT, HH);
    hipLaunchKernelGGL(k_gemm_wmma, dim3(1, NTOT / 16), blk, 0, stream,
                       x_bf, HH, w_outT, HH, b_out, (const float*)nullptr,
                       (float*)d_out, ONF, ONF, HH, 0);
}
